// AttentionPairBias_76716705841244
// MI455X (gfx1250) — matrix operand order repaired and checked by reading, not compiled, not run on hardware
//
#include <hip/hip_runtime.h>
#include <hip/hip_bf16.h>

typedef __attribute__((ext_vector_type(16))) __bf16 v16bf;
typedef __attribute__((ext_vector_type(8)))  float  v8f;
typedef unsigned int u32;
typedef __attribute__((ext_vector_type(4))) u32 u32x4;
typedef __attribute__((ext_vector_type(8))) u32 u32x8;

#define CS 1024
#define HEADS 16
#define HD 64

// pack two f32 into one u32 of bf16 pairs (lowers to v_cvt_pk_bf16_f32)
__device__ inline u32 pk_bf16(float a, float b) {
    union { __bf16 h[2]; u32 w; } u;
    u.h[0] = (__bf16)a; u.h[1] = (__bf16)b;
    return u.w;
}

// read one 32B WMMA fragment (16 bf16) from LDS: two ds_load_b128
__device__ inline v16bf frag_from_lds(const u32* p) {
    union { uint4 q[2]; v16bf f; } u;
    u.q[0] = *(const uint4*)(p);
    u.q[1] = *(const uint4*)(p + 4);
    return u.f;
}

// ---------------------------------------------------------------------------
// Fused LayerNorm(c_z=128) + Linear(128 -> 16 heads) over z (B,N,N,128).
// Block = 256 threads processes 8 consecutive 16-pair tiles (128 pairs):
//   - Wz B-fragments staged ONCE per block into LDS (swizzled packed bf16),
//     then preloaded into registers by every wave
//   - per tile: each wave computes LN stats for 2 pairs, writes normalized
//     values into a double-buffered A-fragment-swizzled LDS tile
//   - WMMA duty rotates across waves (tile t -> wave t&7), overlapping the
//     projection + stores of tile t with staging of tile t+1
// This kernel streams z exactly once (604 MB) and dominates total runtime.
// ---------------------------------------------------------------------------
#define TPB 8   // tiles per block

__global__ __launch_bounds__(256) void pair_bias_kernel(
    const float* __restrict__ z, const float* __restrict__ ln_g,
    const float* __restrict__ ln_b, const float* __restrict__ Wz,
    float* __restrict__ bias, int B, int N)
{
    __shared__ u32 Azn[2 * 4 * 32 * 8];   // double-buffered zn tile (8 KB)
    __shared__ u32 WzF[4 * 32 * 8];       // Wz B-fragments          (4 KB)

    int tid  = threadIdx.x;
    int wave = tid >> 5;
    int lane = tid & 31;
    int half = lane >> 4;
    int lr   = lane & 15;

    long pbase = (long)blockIdx.x * (16 * TPB);   // 128 consecutive pairs
    int  jb = (int)(pbase % N);                   // 128 | 768 -> same (b,i) row
    long t1 = pbase / N;
    int  i  = (int)(t1 % N);
    int  b  = (int)(t1 / N);

    // ---- stage Wz fragments into LDS once (cooperative, 4 u32/thread) ----
    #pragma unroll
    for (int it = 0; it < 4; ++it) {
        int u = tid * 4 + it;            // 0..1023
        int ckk = u >> 8;
        int ln  = (u >> 3) & 31;
        int ew  = u & 7;
        int ch  = ckk * 32 + (ln >> 4) * 16 + ew * 2;
        WzF[u] = pk_bf16(Wz[ch * 16 + (ln & 15)], Wz[(ch + 1) * 16 + (ln & 15)]);
    }
    __syncthreads();
    // every wave holds the 4 Wz B-fragments in registers
    v16bf wfr[4];
    #pragma unroll
    for (int q = 0; q < 4; ++q) wfr[q] = frag_from_lds(&WzF[q * 256 + lane * 8]);

    // per-lane LayerNorm gain/bias for channels c..c+3
    int c = lane * 4;
    float4 g4 = *(const float4*)(ln_g + c);
    float4 b4 = *(const float4*)(ln_b + c);

    // swizzle target for this lane's 4 channels (one (kk,h) run, 2 u32)
    int kk = lane >> 3;
    int cm = (lane & 7) * 4;
    int hh = cm >> 4;
    int e0 = cm & 15;

    for (int t = 0; t < TPB; ++t) {
        u32* buf = &Azn[(t & 1) * 1024];

        // ---- stage tile t: LN stats + swizzled bf16 write (all waves) ----
        #pragma unroll
        for (int pp = 0; pp < 2; ++pp) {
            int row = wave + pp * 8;
            const float* zp = z + (pbase + t * 16 + row) * 128;
            __builtin_prefetch(zp + 16 * 128, 0, 1);
            float4 zv = ((const float4*)zp)[lane];

            float s1 = zv.x + zv.y + zv.z + zv.w;
            float s2 = zv.x * zv.x + zv.y * zv.y + zv.z * zv.z + zv.w * zv.w;
            #pragma unroll
            for (int m = 16; m >= 1; m >>= 1) {
                s1 += __shfl_xor(s1, m);
                s2 += __shfl_xor(s2, m);
            }
            float mu   = s1 * (1.0f / 128.0f);
            float var  = s2 * (1.0f / 128.0f) - mu * mu;
            float rstd = rsqrtf(var + 1e-5f);

            float zn0 = (zv.x - mu) * rstd * g4.x + b4.x;
            float zn1 = (zv.y - mu) * rstd * g4.y + b4.y;
            float zn2 = (zv.z - mu) * rstd * g4.z + b4.z;
            float zn3 = (zv.w - mu) * rstd * g4.w + b4.w;

            u32* dst = &buf[(kk * 32 + hh * 16 + row) * 8 + (e0 >> 1)];
            dst[0] = pk_bf16(zn0, zn1);
            dst[1] = pk_bf16(zn2, zn3);
        }
        __syncthreads();

        // ---- projection: rotates across waves; others run ahead to t+1 ----
        if (wave == (t & 7)) {
            v8f acc = {};
            #pragma unroll
            for (int q = 0; q < 4; ++q) {
                v16bf af = frag_from_lds(&buf[(q * 32 + lane) * 8]);
                acc = __builtin_amdgcn_wmma_f32_16x16x32_bf16(
                    false, af, false, wfr[q], (short)0, acc, false, false);
            }
            int j0 = jb + t * 16;
            // C layout: VGPR r holds pair-row r+8*half, col = head = lane%16
            #pragma unroll
            for (int r = 0; r < 8; ++r) {
                bias[(((size_t)b * HEADS + lr) * N + i) * N + j0 + r + 8 * half]
                    = acc[r];
            }
        }
    }
}

// ---------------------------------------------------------------------------
// GEMM  C(M,N) = act( [A (.* A2)] @ W + bias )  via bf16 WMMA, f32 accumulate.
// Block = 256 threads (8 waves), tile 64x128x32.  LDS holds packed-bf16 tiles
// pre-swizzled into WMMA fragment order: conversions happen ONCE at staging
// and each fragment build is two contiguous ds_load_b128 (zero inner VALU).
// ---------------------------------------------------------------------------
#define BM 64
#define BN 128
#define BKK 32

template <int HAS_A2, int HAS_BIAS, int ACT>
__global__ __launch_bounds__(256) void gemm_bf16_kernel(
    const float* __restrict__ A, const float* __restrict__ A2,
    const float* __restrict__ W, const float* __restrict__ bias,
    float* __restrict__ C, int M, int N, int K)
{
    __shared__ u32 AswU[4 * 32 * 8];   // 4 KB
    __shared__ u32 BswU[8 * 32 * 8];   // 8 KB

    int tid  = threadIdx.x;
    int lane = tid & 31;
    int wave = tid >> 5;
    int wmoff = (wave & 3) * 16;
    int wnoff = (wave >> 2) * 64;
    int m0 = blockIdx.x * BM;
    int n0 = blockIdx.y * BN;
    int half = lane >> 4;
    int lr   = lane & 15;

    v8f acc[4] = {};

    for (int k0 = 0; k0 < K; k0 += BKK) {
        // ---- stage A (64x32 f32 -> swizzled packed bf16) ----
        #pragma unroll
        for (int it = 0; it < 4; ++it) {
            int fp = it * 256 + tid;
            int r  = fp >> 4;
            int cc = (fp & 15) * 2;
            size_t ga = (size_t)(m0 + r) * K + k0 + cc;
            float2 av = *(const float2*)(A + ga);
            if constexpr (HAS_A2) {
                float2 a2 = *(const float2*)(A2 + ga);
                av.x *= a2.x; av.y *= a2.y;
            }
            int h  = cc >> 4;
            int e  = cc & 15;
            AswU[((r >> 4) * 32 + h * 16 + (r & 15)) * 8 + (e >> 1)] =
                pk_bf16(av.x, av.y);
        }
        // ---- stage B (32x128 f32 -> swizzled packed bf16, pack along K) ----
        #pragma unroll
        for (int it = 0; it < 8; ++it) {
            int fl = it * 256 + tid;
            int cc = fl & 127;
            int r  = (fl >> 7) * 2;
            float b0v = W[(size_t)(k0 + r)     * N + n0 + cc];
            float b1v = W[(size_t)(k0 + r + 1) * N + n0 + cc];
            int g  = cc >> 6;
            int nf = (cc >> 4) & 3;
            int ln = (r >> 4) * 16 + (cc & 15);
            BswU[((g * 4 + nf) * 32 + ln) * 8 + ((r & 15) >> 1)] = pk_bf16(b0v, b1v);
        }
        __syncthreads();

        v16bf af = frag_from_lds(&AswU[((wmoff >> 4) * 32 + lane) * 8]);
        #pragma unroll
        for (int nf = 0; nf < 4; ++nf) {
            v16bf bfr = frag_from_lds(&BswU[(((wnoff >> 6) * 4 + nf) * 32 + lane) * 8]);
            acc[nf] = __builtin_amdgcn_wmma_f32_16x16x32_bf16(
                false, af, false, bfr, (short)0, acc[nf], false, false);
        }
        __syncthreads();
    }

    #pragma unroll
    for (int nf = 0; nf < 4; ++nf) {
        #pragma unroll
        for (int r = 0; r < 8; ++r) {
            int row = m0 + wmoff + r + 8 * half;
            int col = n0 + wnoff + nf * 16 + lr;
            float v = acc[nf][r];
            if constexpr (HAS_BIAS) v += bias[col];
            if constexpr (ACT == 1) v = 1.0f / (1.0f + __expf(-v));
            C[(size_t)row * N + col] = v;
        }
    }
}

// ---------------------------------------------------------------------------
// Flash-style attention per (b, h, 16-row i-tile): one wave per block.
// S = (q k^T)/8 + bias + mask, online softmax, O = P v.
//  - K tile staged as packed bf16 (fragment reads = 2x ds_load_b128, no cvt)
//  - bias tile fetched by the Tensor Data Mover (tensor_load_to_lds + wait)
//  - P written to LDS directly in A-fragment-swizzled bf16 layout
// ---------------------------------------------------------------------------
#define KSTR 36   // u32 stride of packed-bf16 k rows (64 bf16 + pad)
#define KVS  68   // f32 stride of v rows

__global__ __launch_bounds__(32) void attn_kernel(
    const float* __restrict__ q, const float* __restrict__ k,
    const float* __restrict__ v, const float* __restrict__ bias,
    const float* __restrict__ mask, float* __restrict__ o, int N)
{
    __shared__ u32   sku[32 * KSTR];
    __shared__ float sv [32 * KVS];
    __shared__ u32   spu[32 * 8];
    __shared__ float sbias[16 * 32];

    int lane = threadIdx.x;
    int half = lane >> 4;
    int lr   = lane & 15;
    int i0 = blockIdx.x * 16;
    int h  = blockIdx.y;
    int b  = blockIdx.z;

    // q A-fragments for dk = 0, 32 (K-dim = head_dim 64), built once
    v16bf qf[2];
    #pragma unroll
    for (int kk = 0; kk < 2; ++kk) {
        const float* qr = q + ((size_t)b * N + i0 + lr) * CS + h * HD + kk * 32;
        int cb = half * 16;
        #pragma unroll
        for (int e = 0; e < 8; ++e) qf[kk][e]     = (__bf16)qr[cb + e];
        #pragma unroll
        for (int e = 0; e < 8; ++e) qf[kk][8 + e] = (__bf16)qr[cb + 8 + e];
    }

    v8f oacc[4] = {};
    float mrow[8], lrow[8];
    #pragma unroll
    for (int r = 0; r < 8; ++r) { mrow[r] = -3.0e38f; lrow[r] = 0.0f; }

    const float scale = 0.125f;
    size_t bb = ((size_t)b * HEADS + h) * N;

    // invariant D# group 1: data_size=4B, tile 32x16, dims/stride = N
    u32x8 g1;
    g1[0] = (2u << 16);
    g1[1] = ((u32)N & 0xFFFFu) << 16;
    g1[2] = (((u32)N >> 16) & 0xFFFFu) | (((u32)N & 0xFFFFu) << 16);
    g1[3] = (((u32)N >> 16) & 0xFFFFu) | (32u << 16);
    g1[4] = 16u;
    g1[5] = (u32)N;
    g1[6] = 0u;
    g1[7] = 0u;
    u32 lds_bias = (u32)(uintptr_t)sbias;

    for (int j0 = 0; j0 < N; j0 += 32) {
        // issue TDM load of the 16x32 bias tile for this step
        {
            unsigned long long va =
                (unsigned long long)(uintptr_t)(bias + (bb + i0) * N + j0);
            u32x4 g0;
            g0[0] = 1u;
            g0[1] = lds_bias;
            g0[2] = (u32)va;
            g0[3] = ((u32)(va >> 32) & 0x01FFFFFFu) | (2u << 30);
            asm volatile("tensor_load_to_lds %0, %1" :: "s"(g0), "s"(g1) : "memory");
        }

        // stage K (packed bf16) and V (f32) tiles, coalesced float4
        #pragma unroll
        for (int t = 0; t < 16; ++t) {
            int flat = t * 128 + lane * 4;
            int r = flat >> 6, cc = flat & 63;
            size_t g = ((size_t)b * N + j0 + r) * CS + h * HD + cc;
            float4 k4 = *(const float4*)(k + g);
            float4 v4 = *(const float4*)(v + g);
            sku[r * KSTR + (cc >> 1)]     = pk_bf16(k4.x, k4.y);
            sku[r * KSTR + (cc >> 1) + 1] = pk_bf16(k4.z, k4.w);
            *(float4*)&sv[r * KVS + cc] = v4;
            if (j0 + 32 < N)
                __builtin_prefetch(k + g + (size_t)32 * CS, 0, 1);
        }
        asm volatile("s_wait_dscnt 0x0" ::: "memory");

        // S tiles: columns j0..j0+15 (c0) and j0+16..j0+31 (c1)
        v8f c0 = {}, c1 = {};
        #pragma unroll
        for (int kk = 0; kk < 2; ++kk) {
            int du = (kk * 32 + half * 16) >> 1;
            v16bf b0 = frag_from_lds(&sku[lr * KSTR + du]);
            v16bf b1 = frag_from_lds(&sku[(16 + lr) * KSTR + du]);
            c0 = __builtin_amdgcn_wmma_f32_16x16x32_bf16(
                false, qf[kk], false, b0, (short)0, c0, false, false);
            c1 = __builtin_amdgcn_wmma_f32_16x16x32_bf16(
                false, qf[kk], false, b1, (short)0, c1, false, false);
        }

        __builtin_amdgcn_s_wait_tensorcnt(0);

        // online softmax over the 32 new columns
        float p0[8], p1[8], esc[8];
        int c0j = j0 + lr, c1j = j0 + 16 + lr;
        float msk0 = (1.0f - mask[(size_t)b * N + c0j]) * -1000000.0f;
        float msk1 = (1.0f - mask[(size_t)b * N + c1j]) * -1000000.0f;
        #pragma unroll
        for (int r = 0; r < 8; ++r) {
            int rowp = r + 8 * half;
            float s0 = c0[r] * scale + sbias[rowp * 32 + lr]      + msk0;
            float s1 = c1[r] * scale + sbias[rowp * 32 + 16 + lr] + msk1;
            float tm = fmaxf(s0, s1);
            #pragma unroll
            for (int m = 8; m >= 1; m >>= 1) tm = fmaxf(tm, __shfl_xor(tm, m));
            float mnew = fmaxf(mrow[r], tm);
            float e    = __expf(mrow[r] - mnew);
            mrow[r] = mnew;
            float e0 = __expf(s0 - mnew);
            float e1 = __expf(s1 - mnew);
            float rs = e0 + e1;
            #pragma unroll
            for (int m = 8; m >= 1; m >>= 1) rs += __shfl_xor(rs, m);
            lrow[r] = lrow[r] * e + rs;
            p0[r] = e0; p1[r] = e1; esc[r] = e;
        }
        #pragma unroll
        for (int nf = 0; nf < 4; ++nf)
            #pragma unroll
            for (int r = 0; r < 8; ++r) oacc[nf][r] *= esc[r];

        // scatter P directly into A-fragment-swizzled bf16 layout
        {
            __bf16* spb = (__bf16*)spu;
            int lbase = (lr >> 3) * 16;
            int eb    = lr & 7;
            #pragma unroll
            for (int r = 0; r < 8; ++r) {
                int rowp = r + 8 * half;
                spb[(lbase + rowp) * 16 + eb]     = (__bf16)p0[r];
                spb[(lbase + rowp) * 16 + eb + 8] = (__bf16)p1[r];
            }
        }
        asm volatile("s_wait_dscnt 0x0" ::: "memory");
        v16bf pf = frag_from_lds(&spu[lane * 8]);

        // O += P (16x32) @ V (32x64)
        #pragma unroll
        for (int nf = 0; nf < 4; ++nf) {
            v16bf vf;
            #pragma unroll
            for (int e = 0; e < 16; ++e) {
                int jj = half * 8 + (e & 7) + ((e >> 3) << 4);
                vf[e] = (__bf16)sv[jj * KVS + nf * 16 + lr];
            }
            oacc[nf] = __builtin_amdgcn_wmma_f32_16x16x32_bf16(
                false, pf, false, vf, (short)0, oacc[nf], false, false);
        }
    }

    #pragma unroll
    for (int nf = 0; nf < 4; ++nf) {
        #pragma unroll
        for (int r = 0; r < 8; ++r) {
            int row = i0 + r + 8 * half;
            int col = h * HD + nf * 16 + lr;
            o[((size_t)b * N + row) * CS + col] = oacc[nf][r] / lrow[r];
        }
    }
}

// ---------------------------------------------------------------------------
extern "C" void kernel_launch(void* const* d_in, const int* in_sizes, int n_in,
                              void* d_out, int out_size, void* d_ws, size_t ws_size,
                              hipStream_t stream) {
    const float* s    = (const float*)d_in[0];
    const float* z    = (const float*)d_in[1];
    const float* mask = (const float*)d_in[2];
    const float* k_in = (const float*)d_in[3];
    const float* Wq   = (const float*)d_in[4];
    const float* bq   = (const float*)d_in[5];
    const float* Wk   = (const float*)d_in[6];
    const float* Wv   = (const float*)d_in[7];
    const float* Wg   = (const float*)d_in[8];
    const float* ln_g = (const float*)d_in[9];
    const float* ln_b = (const float*)d_in[10];
    const float* Wz   = (const float*)d_in[11];
    const float* Wo   = (const float*)d_in[12];

    const int B = 2, N = 768;
    const int M = B * N;

    float* ws   = (float*)d_ws;
    size_t mat  = (size_t)M * CS;
    float* qb   = ws;
    float* kb   = qb + mat;
    float* vb   = kb + mat;
    float* gb   = vb + mat;
    float* ob   = gb + mat;
    float* bias = ob + mat;              // B*H*N*N floats

    // 1) fused LayerNorm + pair-bias projection via WMMA (streams z once)
    {
        long blocks = (long)B * N * N / (16 * TPB);
        pair_bias_kernel<<<(unsigned)blocks, 256, 0, stream>>>(
            z, ln_g, ln_b, Wz, bias, B, N);
    }

    // 2) input projections q/k/v/g (bf16 WMMA GEMMs)
    dim3 gg(M / BM, CS / BN);
    gemm_bf16_kernel<0,1,0><<<gg, 256, 0, stream>>>(s,    nullptr, Wq, bq,      qb, M, CS, CS);
    gemm_bf16_kernel<0,0,0><<<gg, 256, 0, stream>>>(k_in, nullptr, Wk, nullptr, kb, M, CS, CS);
    gemm_bf16_kernel<0,0,0><<<gg, 256, 0, stream>>>(k_in, nullptr, Wv, nullptr, vb, M, CS, CS);
    gemm_bf16_kernel<0,0,1><<<gg, 256, 0, stream>>>(s,    nullptr, Wg, nullptr, gb, M, CS, CS);

    // 3) biased attention with online softmax (WMMA + TDM bias tiles)
    attn_kernel<<<dim3(N / 16, HEADS, B), 32, 0, stream>>>(qb, kb, vb, bias, mask, ob, N);

    // 4) out = (g .* o) @ Wo
    gemm_bf16_kernel<1,0,0><<<gg, 256, 0, stream>>>(ob, gb, Wo, nullptr, (float*)d_out, M, CS, CS);
}
